// SparseAttentionWithVisualization_69853348102278
// MI455X (gfx1250) — matrix operand order repaired and checked by reading, not compiled, not run on hardware
//
#include <hip/hip_runtime.h>

// ---------------------------------------------------------------------------
// Sparse attention with lightning indexer, CDNA5 (gfx1250) implementation.
// Dense projections: v_wmma_f32_16x16x32_bf16 with double-buffered
// global_load_async_to_lds_b128 staging (ASYNCcnt pipeline). Data-dependent
// stages (radix top-k, gathered attention) are VALU wave32 code.
// Workspace requirement: ~163 MB (see layout in kernel_launch).
// ---------------------------------------------------------------------------

typedef unsigned short u16;
typedef unsigned int   u32;
typedef __attribute__((ext_vector_type(16))) __bf16 v16bf;
typedef __attribute__((ext_vector_type(8)))  float  v8f;

#define DEVFN static __device__ __forceinline__

constexpr int B_    = 2;
constexpr int S_    = 4096;
constexpr int D_    = 1024;
constexpr int BS_   = B_ * S_;          // 8192 rows
constexpr int QKVN  = 3 * D_;           // 3072
constexpr int HIDX  = 4;
constexpr int DIDX  = 64;
constexpr int TOPK  = 256;

// ---------------------------------------------------------------------------
// f32 -> bf16 (round to nearest even)
// ---------------------------------------------------------------------------
DEVFN u16 f2b(float f) {
    u32 u = __float_as_uint(f);
    return (u16)((u + 0x7FFFu + ((u >> 16) & 1u)) >> 16);
}

__global__ void cvt_f32_bf16_kernel(const float* __restrict__ src,
                                    u16* __restrict__ dst, size_t n) {
    size_t i = (size_t)blockIdx.x * blockDim.x + threadIdx.x;
    if (i < n) dst[i] = f2b(src[i]);
}

// ---------------------------------------------------------------------------
// CDNA5 async global->LDS staging (ASYNCcnt).
// the LDS address comes from VGPR vdst; low 32 bits of a generic shared pointer are
// the in-allocation LDS byte offset, which is exactly what VDST wants.
// ---------------------------------------------------------------------------
DEVFN void async_load_b128(void* lds_dst, const void* gaddr) {
    u32 lds_off = (u32)(size_t)lds_dst;
    asm volatile("global_load_async_to_lds_b128 %0, %1, off"
                 :: "v"(lds_off), "v"(gaddr)
                 : "memory");
}
DEVFN void wait_async0() { asm volatile("s_wait_asynccnt 0x0" ::: "memory"); }
DEVFN void wait_async4() { asm volatile("s_wait_asynccnt 0x4" ::: "memory"); }

// ---------------------------------------------------------------------------
// Tiled bf16 WMMA GEMM:  C[M,N] = A[M,K] * W[N,K]^T + bias[N]
// 256 threads (8 waves). BM=BN=128, BK=32. Double-buffered async staging.
// Each wave: 32 rows x 64 cols -> 2x4 tiles of 16x16, f32 accumulators.
//
// Requirements (all call sites obey them):
//   * M % 128 == 0, K % 32 == 0  -> no load/store guards on M or K.
//   * W over-read: rows [N, 128*ceil(N/128)) are READ (never stored); the
//     workspace layout guarantees those bytes are mapped. WMMA columns are
//     separable, so garbage there only affects never-stored C columns.
//   * NGUARD=true only adds the col<N store guard (N=64 / N=4 GEMMs).
// ---------------------------------------------------------------------------
constexpr int BM = 128, BN = 128, BK = 32;
constexpr int LDST = BK + 8;   // ushort stride 40 -> 80B rows, 16B aligned

DEVFN v16bf load_fragA(const u16 sA[][LDST], int mBase, int lane) {
    // 16-bit A matrix 16x32: lane%16 -> M, lane/16=sel, K in {sel*8..+7, 16+sel*8..+7}
    int m   = mBase + (lane & 15);
    int sel = lane >> 4;
    union { uint4 q[2]; v16bf v; } f;
    f.q[0] = *(const uint4*)&sA[m][sel * 8];
    f.q[1] = *(const uint4*)&sA[m][16 + sel * 8];
    return f.v;
}

DEVFN v16bf load_fragB(const u16 sB[][LDST], int nBase, int lane) {
    // 16-bit B matrix 32x16: lane%16 -> N, lane/16=kh, K in {kh*8..+7, 16+kh*8..+7}
    int n  = nBase + (lane & 15);
    int kh = lane >> 4;
    union { uint4 q[2]; v16bf v; } f;
    f.q[0] = *(const uint4*)&sB[n][kh * 8];
    f.q[1] = *(const uint4*)&sB[n][16 + kh * 8];
    return f.v;
}

template <bool NGUARD>
__global__ __launch_bounds__(256)
void gemm_bf16_wmma_kernel(const u16* __restrict__ A,      // M x K bf16
                           const u16* __restrict__ W,      // N x K bf16
                           const float* __restrict__ bias, // N (or null)
                           float* __restrict__ C,          // M x N f32
                           int M, int N, int K) {
    __shared__ __align__(16) u16 sA[2][BM][LDST];
    __shared__ __align__(16) u16 sB[2][BN][LDST];

    const int tid  = threadIdx.x;
    const int lane = tid & 31;
    const int wave = tid >> 5;
    const int bm   = blockIdx.y * BM;
    const int bn   = blockIdx.x * BN;
    const int waveM = (wave & 3) * 32;   // 4 waves tile M
    const int waveN = (wave >> 2) * 64;  // 2 waves tile N

    // Per-thread staging coordinates: 2 chunks of 8 bf16 (16B) per matrix.
    const int r0 = (tid * 2) >> 2;            // row of chunk 0
    const int c0 = ((tid * 2) & 3) * 8;       // col of chunk 0
    const int r1 = (tid * 2 + 1) >> 2;
    const int c1 = ((tid * 2 + 1) & 3) * 8;

    // Issue the 4 async (per-wave) staging loads for one K-tile into stage s.
    auto issue = [&](int s, int k0) {
        async_load_b128(&sA[s][r0][c0], A + (size_t)(bm + r0) * K + k0 + c0);
        async_load_b128(&sA[s][r1][c1], A + (size_t)(bm + r1) * K + k0 + c1);
        async_load_b128(&sB[s][r0][c0], W + (size_t)(bn + r0) * K + k0 + c0);
        async_load_b128(&sB[s][r1][c1], W + (size_t)(bn + r1) * K + k0 + c1);
    };

    v8f acc[2][4];
    v8f zero = 0.0f;
#pragma unroll
    for (int i = 0; i < 2; ++i)
#pragma unroll
        for (int j = 0; j < 4; ++j) acc[i][j] = zero;

    const int T = K / BK;
    issue(0, 0);
    for (int i = 0; i < T; ++i) {
        const int cur = i & 1;
        if (i + 1 < T) {
            issue(cur ^ 1, (i + 1) * BK);
            wait_async4();                 // tile i landed; tile i+1 in flight
        } else {
            wait_async0();
        }
        __syncthreads();

        v16bf af[2], bfm[4];
#pragma unroll
        for (int ii = 0; ii < 2; ++ii) af[ii]  = load_fragA(sA[cur], waveM + ii * 16, lane);
#pragma unroll
        for (int jj = 0; jj < 4; ++jj) bfm[jj] = load_fragB(sB[cur], waveN + jj * 16, lane);
#pragma unroll
        for (int ii = 0; ii < 2; ++ii)
#pragma unroll
            for (int jj = 0; jj < 4; ++jj)
                acc[ii][jj] = __builtin_amdgcn_wmma_f32_16x16x32_bf16(
                    false, af[ii], false, bfm[jj], (short)0, acc[ii][jj], false, false);
        __syncthreads();   // all reads of stage `cur` done before tile i+2 lands
    }

    // Store: C/D layout -> vgpr r holds row (base + 8*(lane/16) + r), col lane%16.
    const int hi = lane >> 4;
#pragma unroll
    for (int i = 0; i < 2; ++i) {
#pragma unroll
        for (int j = 0; j < 4; ++j) {
            int col = bn + waveN + j * 16 + (lane & 15);
            if (NGUARD && col >= N) continue;
            float bv = bias ? bias[col] : 0.0f;
#pragma unroll
            for (int r = 0; r < 8; ++r) {
                int row = bm + waveM + i * 16 + hi * 8 + r;
                C[(size_t)row * N + col] = acc[i][j][r] + bv;
            }
        }
    }
}

// ---------------------------------------------------------------------------
// RoPE over full D=1024 (half = 512), applied in-place to Q and K regions
// of the packed qkv buffer (row stride 3072).
// ---------------------------------------------------------------------------
__global__ void rope_kernel(float* __restrict__ qkv) {
    int idx = blockIdx.x * blockDim.x + threadIdx.x;   // over B*S*512
    constexpr int HALF = D_ / 2;
    if (idx >= BS_ * HALF) return;
    int d   = idx % HALF;
    int pos = (idx / HALF) % S_;
    int bs  = idx / HALF;                               // b*S + pos
    float freq = __expf(-9.210340372f * (float)d / (float)HALF); // 10000^{-d/half}
    float ang  = (float)pos * freq;
    float sn, cs;
    __sincosf(ang, &sn, &cs);
    float* row = qkv + (size_t)bs * QKVN;
    float q1 = row[d], q2 = row[d + HALF];
    row[d]        = q1 * cs - q2 * sn;
    row[d + HALF] = q1 * sn + q2 * cs;
    float k1 = row[D_ + d], k2 = row[D_ + d + HALF];
    row[D_ + d]        = k1 * cs - k2 * sn;
    row[D_ + d + HALF] = k1 * sn + k2 * cs;
}

// ---------------------------------------------------------------------------
// Per-query index scores + exact top-256 via 4-pass LDS radix select.
// One block (256 threads) per (b, q). Scores live in LDS as order-preserving
// u32 keys (16 KB), well inside the 320 KB / WGP budget.
// ---------------------------------------------------------------------------
__global__ __launch_bounds__(256)
void index_topk_kernel(const float* __restrict__ iq,  // [BS,256]
                       const float* __restrict__ ik,  // [BS,64]
                       const float* __restrict__ iw,  // [BS,4]
                       int* __restrict__ topidx) {    // [BS,256]
    __shared__ u32   ukey[S_];
    __shared__ float s_iq[HIDX * DIDX];
    __shared__ float s_iw[HIDX];
    __shared__ u32   hist[256];
    __shared__ u32   sc[2];
    __shared__ u32   outcnt, tiecnt;

    const int bq  = blockIdx.x;         // b*S + q
    const int b   = bq / S_;
    const int tid = threadIdx.x;

    s_iq[tid] = iq[(size_t)bq * (HIDX * DIDX) + tid];
    if (tid < HIDX) s_iw[tid] = iw[(size_t)bq * HIDX + tid];
    if (tid == 0) { outcnt = 0u; tiecnt = 0u; }
    __syncthreads();

    // Scores: score(k) = sum_h relu(iq_h . ik_k) * iw_h
    const float* ikb = ik + (size_t)b * S_ * DIDX;
    for (int k = tid; k < S_; k += 256) {
        const float* kr = ikb + (size_t)k * DIDX;
        float a0 = 0.f, a1 = 0.f, a2 = 0.f, a3 = 0.f;
#pragma unroll 8
        for (int d = 0; d < DIDX; ++d) {
            float kd = kr[d];
            a0 += s_iq[0 * DIDX + d] * kd;
            a1 += s_iq[1 * DIDX + d] * kd;
            a2 += s_iq[2 * DIDX + d] * kd;
            a3 += s_iq[3 * DIDX + d] * kd;
        }
        float sck = fmaxf(a0, 0.f) * s_iw[0] + fmaxf(a1, 0.f) * s_iw[1] +
                    fmaxf(a2, 0.f) * s_iw[2] + fmaxf(a3, 0.f) * s_iw[3];
        u32 u = __float_as_uint(sck);
        u = (u & 0x80000000u) ? ~u : (u | 0x80000000u);   // order-preserving
        ukey[k] = u;
    }
    __syncthreads();

    // 4-pass radix select (MSB->LSB), descending, exact 256.
    u32 prefix = 0u;
    int needed = TOPK;
    for (int pass = 3; pass >= 0; --pass) {
        const int shift = pass * 8;
        hist[tid] = 0u;
        __syncthreads();
        const u32 himask = (pass == 3) ? 0u : (0xFFFFFFFFu << (shift + 8));
        for (int k = tid; k < S_; k += 256) {
            u32 u = ukey[k];
            if (pass == 3 || (u & himask) == prefix)
                atomicAdd(&hist[(u >> shift) & 0xFFu], 1u);
        }
        __syncthreads();
        if (tid == 0) {
            u32 cum = 0u;
            u32 bsel = 0u;
            for (int bkt = 255; bkt >= 0; --bkt) {
                u32 c = hist[bkt];
                if (cum + c >= (u32)needed) { bsel = (u32)bkt; break; }
                cum += c;
            }
            sc[0] = cum;    // count strictly above selected bucket
            sc[1] = bsel;
        }
        __syncthreads();
        needed -= (int)sc[0];
        prefix |= (sc[1] << shift);
        __syncthreads();
    }

    // Collect: all keys > prefix, plus `needed` ties at == prefix.
    int* outp = topidx + (size_t)bq * TOPK;
    for (int k = tid; k < S_; k += 256) {
        u32 u = ukey[k];
        bool take = (u > prefix);
        if (!take && u == prefix) take = (atomicAdd(&tiecnt, 1u) < (u32)needed);
        if (take) {
            u32 p = atomicAdd(&outcnt, 1u);
            if (p < (u32)TOPK) outp[p] = k;
        }
    }
}

// ---------------------------------------------------------------------------
// Sparse attention over the 256 selected keys. One block per (b, q).
// Thread-per-key QK dots; block softmax (wave32 shuffles); coalesced float4
// P*V accumulation. Output written as bf16 (input to the final WMMA GEMM).
// ---------------------------------------------------------------------------
__global__ __launch_bounds__(256)
void sparse_attn_kernel(const float* __restrict__ qkv,     // [BS,3072] (roped)
                        const int* __restrict__ topidx,    // [BS,256]
                        u16* __restrict__ attnb) {         // [BS,1024] bf16
    __shared__ float qrow[D_];
    __shared__ float prob[TOPK];
    __shared__ int   sidx[TOPK];
    __shared__ float red1[8];
    __shared__ float red2[8];

    const int bq  = blockIdx.x;
    const int b   = bq / S_;
    const int tid = threadIdx.x;

    const float* Qr = qkv + (size_t)bq * QKVN;
    for (int d = tid; d < D_; d += 256) qrow[d] = Qr[d];
    sidx[tid] = topidx[(size_t)bq * TOPK + tid];
    __syncthreads();

    const float* Kbase = qkv + (size_t)b * S_ * QKVN + D_;
    const float* Vbase = qkv + (size_t)b * S_ * QKVN + 2 * D_;

    // logit for my key
    const float* kr = Kbase + (size_t)sidx[tid] * QKVN;
    float acc = 0.f;
#pragma unroll 4
    for (int d = 0; d < D_; d += 4) {
        float4 kv = *(const float4*)(kr + d);
        acc += qrow[d] * kv.x + qrow[d + 1] * kv.y +
               qrow[d + 2] * kv.z + qrow[d + 3] * kv.w;
    }
    float logit = acc * 0.03125f;   // 1/sqrt(1024)

    // block max (wave32 shuffles + LDS)
    float m = logit;
#pragma unroll
    for (int off = 16; off > 0; off >>= 1) m = fmaxf(m, __shfl_xor(m, off));
    if ((tid & 31) == 0) red1[tid >> 5] = m;
    __syncthreads();
    float mg = red1[0];
#pragma unroll
    for (int i = 1; i < 8; ++i) mg = fmaxf(mg, red1[i]);

    float e = __expf(logit - mg);
    float s = e;
#pragma unroll
    for (int off = 16; off > 0; off >>= 1) s += __shfl_xor(s, off);
    if ((tid & 31) == 0) red2[tid >> 5] = s;
    __syncthreads();
    float tot = red2[0];
#pragma unroll
    for (int i = 1; i < 8; ++i) tot += red2[i];
    prob[tid] = e / tot;
    __syncthreads();

    // output: 4 contiguous dims per thread, coalesced across the block
    const int d0 = tid * 4;
    float4 o = make_float4(0.f, 0.f, 0.f, 0.f);
    for (int k = 0; k < TOPK; ++k) {
        float p = prob[k];
        float4 vv = *(const float4*)(Vbase + (size_t)sidx[k] * QKVN + d0);
        o.x += p * vv.x; o.y += p * vv.y; o.z += p * vv.z; o.w += p * vv.w;
    }
    u16* dst = attnb + (size_t)bq * D_ + d0;
    dst[0] = f2b(o.x); dst[1] = f2b(o.y); dst[2] = f2b(o.z); dst[3] = f2b(o.w);
}

// ---------------------------------------------------------------------------
// Host side
// ---------------------------------------------------------------------------
static inline int cdiv(int a, int b) { return (a + b - 1) / b; }

extern "C" void kernel_launch(void* const* d_in, const int* in_sizes, int n_in,
                              void* d_out, int out_size, void* d_ws, size_t ws_size,
                              hipStream_t stream) {
    (void)in_sizes; (void)n_in; (void)out_size; (void)ws_size;

    const float* x     = (const float*)d_in[0];   // [B,S,D]
    const float* w_qkv = (const float*)d_in[1];   // [3D,D]
    const float* b_qkv = (const float*)d_in[2];   // [3D]
    const float* w_o   = (const float*)d_in[3];   // [D,D]
    const float* b_o   = (const float*)d_in[4];   // [D]
    const float* w_iq  = (const float*)d_in[5];   // [256,D]
    const float* b_iq  = (const float*)d_in[6];
    const float* w_ik  = (const float*)d_in[7];   // [64,D]
    const float* b_ik  = (const float*)d_in[8];
    const float* w_iw  = (const float*)d_in[9];   // [4,D]
    const float* b_iw  = (const float*)d_in[10];
    float* out = (float*)d_out;

    // Workspace carve-out (256B aligned slices), total ~163 MB.
    // NOTE: wikb (N=64) and wiwb (N=4) are over-read up to row 127 by the
    // unguarded GEMM staging; they are followed by multi-MB slices, so the
    // over-read stays inside the workspace. A tail pad guarantees slack.
    char* p = (char*)d_ws;
    auto carve = [&](size_t bytes) -> void* {
        void* r = (void*)p;
        p += (bytes + 255) & ~(size_t)255;
        return r;
    };
    u16*   xb    = (u16*)  carve((size_t)BS_ * D_ * 2);        // x bf16
    u16*   wqkvb = (u16*)  carve((size_t)QKVN * D_ * 2);
    u16*   wiqb  = (u16*)  carve((size_t)256 * D_ * 2);
    u16*   wikb  = (u16*)  carve((size_t)64 * D_ * 2);
    u16*   wiwb  = (u16*)  carve((size_t)4 * D_ * 2);
    u16*   wob   = (u16*)  carve((size_t)D_ * D_ * 2);
    float* qkv   = (float*)carve((size_t)BS_ * QKVN * 4);      // 100.7 MB
    float* iqv   = (float*)carve((size_t)BS_ * 256 * 4);
    float* ikv   = (float*)carve((size_t)BS_ * 64 * 4);
    float* iwv   = (float*)carve((size_t)BS_ * 4 * 4);
    int*   topix = (int*)  carve((size_t)BS_ * TOPK * 4);
    u16*   attnb = (u16*)  carve((size_t)BS_ * D_ * 2);
    (void)carve((size_t)256 * 1024);                           // tail pad

    // 1) f32 -> bf16 conversions
    {
        auto cvt = [&](const float* s, u16* d, size_t n) {
            cvt_f32_bf16_kernel<<<(unsigned)((n + 255) / 256), 256, 0, stream>>>(s, d, n);
        };
        cvt(x,     xb,    (size_t)BS_ * D_);
        cvt(w_qkv, wqkvb, (size_t)QKVN * D_);
        cvt(w_iq,  wiqb,  (size_t)256 * D_);
        cvt(w_ik,  wikb,  (size_t)64 * D_);
        cvt(w_iw,  wiwb,  (size_t)4 * D_);
        cvt(w_o,   wob,   (size_t)D_ * D_);
    }

    // 2) Projections on the WMMA path (N%128==0 -> no store guard)
    {
        dim3 blk(256);
        gemm_bf16_wmma_kernel<false><<<dim3(cdiv(QKVN, BN), cdiv(BS_, BM)), blk, 0, stream>>>(
            xb, wqkvb, b_qkv, qkv, BS_, QKVN, D_);
        gemm_bf16_wmma_kernel<false><<<dim3(cdiv(256, BN), cdiv(BS_, BM)), blk, 0, stream>>>(
            xb, wiqb, b_iq, iqv, BS_, 256, D_);
        gemm_bf16_wmma_kernel<true><<<dim3(cdiv(64, BN), cdiv(BS_, BM)), blk, 0, stream>>>(
            xb, wikb, b_ik, ikv, BS_, 64, D_);
        gemm_bf16_wmma_kernel<true><<<dim3(cdiv(4, BN), cdiv(BS_, BM)), blk, 0, stream>>>(
            xb, wiwb, b_iw, iwv, BS_, 4, D_);
    }

    // 3) RoPE on Q and K (in place)
    {
        int n = BS_ * (D_ / 2);
        rope_kernel<<<cdiv(n, 256), 256, 0, stream>>>(qkv);
    }

    // 4) Index scores + exact top-256 per query
    index_topk_kernel<<<BS_, 256, 0, stream>>>(iqv, ikv, iwv, topix);

    // 5) Sparse attention over gathered keys -> bf16 activations
    sparse_attn_kernel<<<BS_, 256, 0, stream>>>(qkv, topix, attnb);

    // 6) Output projection (WMMA) -> f32 result
    gemm_bf16_wmma_kernel<false><<<dim3(cdiv(D_, BN), cdiv(BS_, BM)), 256, 0, stream>>>(
        attnb, wob, b_o, out, BS_, D_, D_);
}
